// LocalGraphAttention_24584392802962
// MI455X (gfx1250) — matrix operand order repaired and checked by reading, not compiled, not run on hardware
//
#include <hip/hip_runtime.h>
#include <hip/hip_bf16.h>

// ---------------------------------------------------------------------------
// LocalGraphAttention for MI455X (gfx1250): f16 WMMA flash attention.
//   D_MODEL=256, N_HEADS=8, D_HEAD=32, B=4, G=2048
// ---------------------------------------------------------------------------

typedef __attribute__((ext_vector_type(16))) _Float16 v16h;
typedef __attribute__((ext_vector_type(8)))  _Float16 v8h;
typedef __attribute__((ext_vector_type(8)))  float    v8f;

#define DMODEL 256
#define NHEADS 8
#define DHEAD  32
#define BATCH  4
#define SEQ    2048
#define QKSCALE 0.17677669529663687f   // 1/sqrt(32)

// ---- WMMA helpers ----------------------------------------------------------

__device__ __forceinline__ v8f wmma16x16x32(v16h a, v16h b, v8f c) {
  // (neg_a, A, neg_b, B, c_mod, C, reuse_a, reuse_b)
  return __builtin_amdgcn_wmma_f32_16x16x32_f16(false, a, false, b, (short)0, c,
                                                false, false);
}

// A-matrix 16x32 f16 layout (ISA 7.12.2): lanes 0-15 hold row (lane&15),
// K in {c0..c0+7, 16+c0..16+c0+7} with c0 = (lane>>4)*8.
__device__ __forceinline__ v16h load_a16(const _Float16* base, int stride, int lane) {
  const int r  = lane & 15;
  const int c0 = (lane >> 4) << 3;
  const _Float16* p = base + r * stride;
  v8h lo = *(const v8h*)(p + c0);
  v8h hi = *(const v8h*)(p + 16 + c0);
  v16h out;
#pragma unroll
  for (int i = 0; i < 8; ++i) { out[i] = lo[i]; out[i + 8] = hi[i]; }
  return out;
}

// B-matrix 32x16 f16: lane = column (lane&15), K in {c0..c0+7, 16+c0..16+c0+7}
// with c0 = (lane>>4)*8. `base` points at the (column-major) operand, i.e. row n
// of the pre-transposed storage.
__device__ __forceinline__ v16h load_b16(const _Float16* base, int stride, int lane) {
  const int n  = lane & 15;
  const int c0 = (lane >> 4) << 3;
  const _Float16* p = base + n * stride;
  v8h lo = *(const v8h*)(p + c0);
  v8h hi = *(const v8h*)(p + 16 + c0);
  v16h out;
#pragma unroll
  for (int i = 0; i < 8; ++i) { out[i] = lo[i]; out[i + 8] = hi[i]; }
  return out;
}

// 16-lane (half-wave) reductions; xor masks 1..8 stay inside each 16-lane group.
__device__ __forceinline__ float hmax16(float v) {
#pragma unroll
  for (int m = 1; m <= 8; m <<= 1) v = fmaxf(v, __shfl_xor(v, m, 32));
  return v;
}
__device__ __forceinline__ float hsum16(float v) {
#pragma unroll
  for (int m = 1; m <= 8; m <<= 1) v += __shfl_xor(v, m, 32);
  return v;
}

// ---- Prep: f32->f16 convert, weight transposes, mask bit-pack --------------

__global__ void lga_prep_convert(const float* __restrict__ x,
                                 const float* __restrict__ Wqkv,
                                 const float* __restrict__ Wout,
                                 _Float16* __restrict__ xh,
                                 _Float16* __restrict__ Wqkvt,
                                 _Float16* __restrict__ Wot) {
  const int total = BATCH * SEQ * DMODEL;          // 2M (largest stream)
  for (int idx = blockIdx.x * blockDim.x + threadIdx.x; idx < total;
       idx += gridDim.x * blockDim.x) {
    xh[idx] = (_Float16)x[idx];
    if (idx < 3 * DMODEL * DMODEL) {               // W_qkv: (256,768) -> (768,256)
      const int j = idx >> 8, k = idx & 255;
      Wqkvt[idx] = (_Float16)Wqkv[k * (3 * DMODEL) + j];
    }
    if (idx < DMODEL * DMODEL) {                   // W_out: (256,256) -> (256,256)^T
      const int j = idx >> 8, k = idx & 255;
      Wot[idx] = (_Float16)Wout[k * DMODEL + j];
    }
  }
}

__global__ void lga_prep_mask(const unsigned char* __restrict__ allow,
                              unsigned* __restrict__ maskbits) {
  const int idx = blockIdx.x * blockDim.x + threadIdx.x;   // one 32-key word
  if (idx >= SEQ * (SEQ / 32)) return;
  const int row = idx >> 6, w = idx & 63;
  const unsigned char* p = allow + row * SEQ + w * 32;
  unsigned word = 0u;
#pragma unroll
  for (int b = 0; b < 32; ++b) word |= (p[b] ? 1u : 0u) << b;
  maskbits[idx] = word;
}

// ---- QKV projection: qkv = xh @ Wqkvt^T + b; scatter into Q / K / V^T ------

__global__ void __launch_bounds__(256)
lga_qkv_proj(const _Float16* __restrict__ xh, const _Float16* __restrict__ Wqkvt,
             const float* __restrict__ bqkv,
             _Float16* __restrict__ Qh, _Float16* __restrict__ Kh,
             _Float16* __restrict__ Vt) {
  const int lane = threadIdx.x & 31;
  const int tile = blockIdx.x * 8 + (threadIdx.x >> 5);  // 24576 tiles exactly
  const int mtile = tile / 48;                           // 512 row tiles
  const int ntile = tile % 48;                           // 48 col tiles (768/16)
  const int jbase = ntile * 16;

  const _Float16* abase = xh + mtile * 16 * DMODEL;
  const _Float16* bbase = Wqkvt + jbase * DMODEL;
  v8f acc = {};
#pragma unroll
  for (int kb = 0; kb < DMODEL / 32; ++kb)
    acc = wmma16x16x32(load_a16(abase + kb * 32, DMODEL, lane),
                       load_b16(bbase + kb * 32, DMODEL, lane), acc);

  const int n = lane & 15, half = lane >> 4;
  const float bj = bqkv[jbase + n];
  const int s    = jbase >> 8;          // 0=Q 1=K 2=V  (wave-uniform)
  const int rem  = jbase & 255;
  const int head = rem >> 5;
  const int dd   = (rem & 31) + n;      // 0..31 within head
#pragma unroll
  for (int r = 0; r < 8; ++r) {
    const int grow = mtile * 16 + r + half * 8;  // 0..8191 = b*G + g
    const int bb = grow >> 11, gg = grow & 2047;
    const int bh = bb * NHEADS + head;
    const float v = acc[r] + bj;
    if (s == 0)
      Qh[(bh * SEQ + gg) * DHEAD + dd] = (_Float16)(v * QKSCALE);
    else if (s == 1)
      Kh[(bh * SEQ + gg) * DHEAD + dd] = (_Float16)v;
    else
      Vt[(bh * DHEAD + dd) * SEQ + gg] = (_Float16)v;
  }
}

// ---- Flash attention: one wave = 16 query rows of one (b,head) -------------

__global__ void __launch_bounds__(256)
lga_attention(const _Float16* __restrict__ Qh, const _Float16* __restrict__ Kh,
              const _Float16* __restrict__ Vt, const unsigned* __restrict__ maskbits,
              _Float16* __restrict__ Oh) {
  __shared__ __align__(16) _Float16 pbuf[8][16][32];   // per-wave P staging (8 KB)

  const int lane  = threadIdx.x & 31;
  const int wid   = threadIdx.x >> 5;
  const int wave  = blockIdx.x * 8 + wid;   // 4096 waves exactly
  const int bh    = wave >> 7;              // 0..31
  const int qtile = wave & 127;
  const int qbase = qtile * 16;
  const int n = lane & 15, half = lane >> 4;

  const _Float16* Qb = Qh + (size_t)bh * SEQ * DHEAD;
  const _Float16* Kb = Kh + (size_t)bh * SEQ * DHEAD;
  const _Float16* Vb = Vt + (size_t)bh * DHEAD * SEQ;

  const v16h qa = load_a16(Qb + qbase * DHEAD, DHEAD, lane);  // 16 q-rows x d=32

  v8f o0 = {}, o1 = {};                 // 16 q-rows x d[0:16], d[16:32]
  float m[8], l[8];
#pragma unroll
  for (int r = 0; r < 8; ++r) { m[r] = -1e30f; l[r] = 0.f; }

  const v8f zero = {};
  for (int chunk = 0; chunk < SEQ / 32; ++chunk) {
    const int kc = chunk * 32;

    // S = Q * K^T for keys [kc, kc+32): one WMMA per 16 keys (K-dim = d = 32)
    v16h kb0 = load_b16(Kb + kc * DHEAD, DHEAD, lane);
    v16h kb1 = load_b16(Kb + (kc + 16) * DHEAD, DHEAD, lane);
    v8f s0 = wmma16x16x32(qa, kb0, zero);
    v8f s1 = wmma16x16x32(qa, kb1, zero);

    if (chunk + 1 < SEQ / 32) {                      // global_prefetch_b8
      __builtin_prefetch(Kb + (kc + 32) * DHEAD, 0, 0);
      __builtin_prefetch(Vb + kc + 32, 0, 0);
    }

    // Online softmax; mask via bit-packed allow words (-1e30 keeps exp finite).
#pragma unroll
    for (int r = 0; r < 8; ++r) {
      const unsigned w = maskbits[(qbase + r + half * 8) * (SEQ / 32) + chunk];
      const float a0 = ((w >> n) & 1u)        ? s0[r] : -1e30f;
      const float a1 = ((w >> (n + 16)) & 1u) ? s1[r] : -1e30f;
      const float mn = fmaxf(m[r], hmax16(fmaxf(a0, a1)));
      const float al = __expf(m[r] - mn);
      const float p0 = __expf(a0 - mn);
      const float p1 = __expf(a1 - mn);
      l[r] = l[r] * al + hsum16(p0 + p1);
      m[r] = mn;
      o0[r] *= al; o1[r] *= al;
      pbuf[wid][r + half * 8][n]      = (_Float16)p0;   // C-layout -> LDS
      pbuf[wid][r + half * 8][n + 16] = (_Float16)p1;
    }

    // Re-read P in A-layout (wave-private LDS; DScnt ordering, no barrier).
    v16h pa = load_a16(&pbuf[wid][0][0], 32, lane);

    // O += P * V : K-dim = 32 keys, two N-halves of d
    v16h vb0 = load_b16(Vb + kc, SEQ, lane);
    v16h vb1 = load_b16(Vb + 16 * SEQ + kc, SEQ, lane);
    o0 = wmma16x16x32(pa, vb0, o0);
    o1 = wmma16x16x32(pa, vb1, o1);
  }

  const int b = bh >> 3, head = bh & 7;
#pragma unroll
  for (int r = 0; r < 8; ++r) {
    const float inv = 1.0f / l[r];
    const int grow = b * SEQ + qbase + r + half * 8;
    Oh[grow * DMODEL + head * DHEAD + n]      = (_Float16)(o0[r] * inv);
    Oh[grow * DMODEL + head * DHEAD + 16 + n] = (_Float16)(o1[r] * inv);
  }
}

// ---- Output projection: y = Oh @ Wot^T + b_out (f32 out) -------------------

__global__ void __launch_bounds__(256)
lga_out_proj(const _Float16* __restrict__ Oh, const _Float16* __restrict__ Wot,
             const float* __restrict__ bout, float* __restrict__ y) {
  const int lane = threadIdx.x & 31;
  const int tile = blockIdx.x * 8 + (threadIdx.x >> 5);  // 8192 tiles exactly
  const int mtile = tile >> 4;
  const int ntile = tile & 15;
  const int jbase = ntile * 16;

  const _Float16* abase = Oh + mtile * 16 * DMODEL;
  const _Float16* bbase = Wot + jbase * DMODEL;
  v8f acc = {};
#pragma unroll
  for (int kb = 0; kb < DMODEL / 32; ++kb)
    acc = wmma16x16x32(load_a16(abase + kb * 32, DMODEL, lane),
                       load_b16(bbase + kb * 32, DMODEL, lane), acc);

  const int n = lane & 15, half = lane >> 4;
  const float bj = bout[jbase + n];
#pragma unroll
  for (int r = 0; r < 8; ++r)
    y[(mtile * 16 + r + half * 8) * DMODEL + jbase + n] = acc[r] + bj;
}

// ---------------------------------------------------------------------------

extern "C" void kernel_launch(void* const* d_in, const int* in_sizes, int n_in,
                              void* d_out, int out_size, void* d_ws, size_t ws_size,
                              hipStream_t stream) {
  const float*         x     = (const float*)d_in[0];
  const unsigned char* allow = (const unsigned char*)d_in[1];  // jnp bool = 1B
  const float*         Wqkv  = (const float*)d_in[2];
  const float*         bqkv  = (const float*)d_in[3];
  const float*         Wout  = (const float*)d_in[4];
  const float*         bout  = (const float*)d_in[5];
  float*               y     = (float*)d_out;

  char* ws = (char*)d_ws;
  size_t off = 0;
  auto alloc = [&](size_t bytes) -> void* {
    void* p = ws + off;
    off = (off + bytes + 255) & ~(size_t)255;
    return p;
  };

  _Float16* xh     = (_Float16*)alloc((size_t)BATCH * SEQ * DMODEL * 2);   // 4 MB
  _Float16* Wqkvt  = (_Float16*)alloc((size_t)3 * DMODEL * DMODEL * 2);    // 384 KB
  _Float16* Wot    = (_Float16*)alloc((size_t)DMODEL * DMODEL * 2);        // 128 KB
  _Float16* Qh     = (_Float16*)alloc((size_t)BATCH * NHEADS * SEQ * DHEAD * 2);
  _Float16* Kh     = (_Float16*)alloc((size_t)BATCH * NHEADS * SEQ * DHEAD * 2);
  _Float16* Vt     = (_Float16*)alloc((size_t)BATCH * NHEADS * DHEAD * SEQ * 2);
  unsigned* mbits  = (unsigned*)alloc((size_t)SEQ * (SEQ / 32) * 4);       // 512 KB
  _Float16* Oh     = xh;  // xh dead after qkv_proj; reuse for attention output

  // 1) prep: converts + transposes + mask bit-pack
  lga_prep_convert<<<2048, 256, 0, stream>>>(x, Wqkv, Wout, xh, Wqkvt, Wot);
  lga_prep_mask<<<(SEQ * (SEQ / 32) + 255) / 256, 256, 0, stream>>>(allow, mbits);

  // 2) QKV projection: 512 * 48 tiles, 8 waves/block
  lga_qkv_proj<<<(512 * 48) / 8, 256, 0, stream>>>(xh, Wqkvt, bqkv, Qh, Kh, Vt);

  // 3) Flash attention: 4 * 8 * 128 waves, 8 waves/block
  lga_attention<<<(BATCH * NHEADS * (SEQ / 16)) / 8, 256, 0, stream>>>(
      Qh, Kh, Vt, mbits, Oh);

  // 4) Output projection: 512 * 16 tiles, 8 waves/block
  lga_out_proj<<<(512 * 16) / 8, 256, 0, stream>>>(Oh, Wot, bout, y);
}
